// MCPRN_37357625541367
// MI455X (gfx1250) — matrix operand order repaired and checked by reading, not compiled, not run on hardware
//
#include <hip/hip_runtime.h>

#define N_ITEMS 100001
#define DIM     128
#define NP      3
#define SLEN    100
#define BATCH   256
#define TAUV    0.1f
#define EPSV    0.01f

typedef __attribute__((ext_vector_type(16))) __bf16 v16bf;
typedef __attribute__((ext_vector_type(8)))  float  v8f;

__device__ __forceinline__ float sigmoidf(float x) { return 1.0f / (1.0f + __expf(-x)); }
__device__ __forceinline__ float tanh_fast(float x) { return 2.0f / (1.0f + __expf(-2.0f * x)) - 1.0f; }
__device__ __forceinline__ float wred32(float v) {
#pragma unroll
  for (int off = 16; off > 0; off >>= 1) v += __shfl_xor(v, off, 32);
  return v;
}

// ---------------------------------------------------------------------------
// Kernel 1: per-item inverse norms + purpose gates g = softmax(item_n·attr_n/τ)
// One wave per item row; 8 waves (256 threads) per block.
// ---------------------------------------------------------------------------
__global__ __launch_bounds__(256) void norm_gate_kernel(
    const float* __restrict__ emb, const float* __restrict__ purpose,
    float* __restrict__ inv_norm, float* __restrict__ g) {
  __shared__ float attr[NP][DIM];
  const int tid = threadIdx.x, wave = tid >> 5, lane = tid & 31;

  if (wave < NP) {  // normalize purpose rows (one wave each)
    float4 v = ((const float4*)(purpose + wave * DIM))[lane];
    float ss = wred32(v.x * v.x + v.y * v.y + v.z * v.z + v.w * v.w);
    float inv = 1.0f / fmaxf(sqrtf(ss), 1e-12f);
    attr[wave][lane * 4 + 0] = v.x * inv;
    attr[wave][lane * 4 + 1] = v.y * inv;
    attr[wave][lane * 4 + 2] = v.z * inv;
    attr[wave][lane * 4 + 3] = v.w * inv;
  }
  __syncthreads();

  const long row = (long)blockIdx.x * 8 + wave;
  if (row >= N_ITEMS) return;

  float4 v = ((const float4*)(emb + (size_t)row * DIM))[lane];
  float inv = 1.0f / fmaxf(sqrtf(wred32(v.x * v.x + v.y * v.y + v.z * v.z + v.w * v.w)), 1e-12f);

  const int k = lane * 4;
  float d[NP];
#pragma unroll
  for (int p = 0; p < NP; ++p) {
    d[p] = wred32(v.x * attr[p][k] + v.y * attr[p][k + 1] +
                  v.z * attr[p][k + 2] + v.w * attr[p][k + 3]);
  }
  if (lane == 0) {
    if (row == 0) {  // reference zeroes emb row 0 and g[0]
      inv_norm[0] = 0.0f;
      g[0] = 0.0f; g[1] = 0.0f; g[2] = 0.0f;
    } else {
      inv_norm[row] = inv;
      float l0 = d[0] * inv / TAUV, l1 = d[1] * inv / TAUV, l2 = d[2] * inv / TAUV;
      float m = fmaxf(l0, fmaxf(l1, l2));
      float e0 = __expf(l0 - m), e1 = __expf(l1 - m), e2 = __expf(l2 - m);
      float rs = 1.0f / (e0 + e1 + e2);
      g[row * 3 + 0] = e0 * rs; g[row * 3 + 1] = e1 * rs; g[row * 3 + 2] = e2 * rs;
    }
  }
}

// ---------------------------------------------------------------------------
// Kernel 2: persistent PSRU recurrence. One workgroup per purpose p.
// h (f32) + Wh (bf16) resident in LDS; gx fused into the same WMMA accumulators
// (A from gathered embedding rows, B from x2h_w streamed through L2).
// ---------------------------------------------------------------------------
#define HP 132  // h LDS pitch (floats) — de-conflicts banks
#define WP 132  // Wh LDS pitch (bf16)

__global__ __launch_bounds__(1024) void rnn_kernel(
    const int* __restrict__ seq, const int* __restrict__ lens,
    const float* __restrict__ emb, const float* __restrict__ x2h,
    const float* __restrict__ h2h, const float* __restrict__ g,
    float* __restrict__ h_last) {
  __shared__ float  h_lds[BATCH * HP];
  __shared__ __bf16 wh_lds[3 * DIM * WP];
  __shared__ int    lens_lds[BATCH];

  const int p = blockIdx.x;
  const int tid = threadIdx.x;

  for (int i = tid; i < BATCH * HP; i += 1024) h_lds[i] = 0.0f;
  const float* whp = h2h + (size_t)p * (3 * DIM) * DIM;
  for (int i = tid; i < 3 * DIM * DIM; i += 1024) {
    int r = i >> 7, c = i & 127;
    wh_lds[r * WP + c] = (__bf16)whp[i];
  }
  if (tid < BATCH) lens_lds[tid] = lens[tid];
  __syncthreads();

  const int wave = tid >> 5, lane = tid & 31;
  const int half = lane >> 4, nn = lane & 15;
  const float* xwp = x2h + (size_t)p * (3 * DIM) * DIM;

  float hnew[4][8];

  for (int t = 0; t < SLEN; ++t) {
    const int* seq_t = seq + t * BATCH;

#pragma unroll
    for (int tt = 0; tt < 4; ++tt) {
      const int tile = wave + tt * 32;
      const int mb = tile >> 3, db = tile & 7;
      v8f acc_ri = {}, acc_ii = {}, acc_xn = {}, acc_hn = {};

      const int mrow = mb * 16 + nn;  // A-matrix row held by this lane
      const int sid = seq_t[mrow];
      const float* xrow = emb + (size_t)sid * DIM;
      const float* hrow = h_lds + mrow * HP;

      for (int kb = 0; kb < 4; ++kb) {
        const int b0 = kb * 32 + half * 16, b1 = b0 + 8;  // A K-mapping
        v16bf a_h, a_x;
#pragma unroll
        for (int j = 0; j < 8; ++j) {
          a_h[j]     = (__bf16)hrow[b0 + j];
          a_h[j + 8] = (__bf16)hrow[b1 + j];
          float x0 = (sid != 0) ? xrow[b0 + j] : 0.0f;
          float x1 = (sid != 0) ? xrow[b1 + j] : 0.0f;
          a_x[j] = (__bf16)x0; a_x[j + 8] = (__bf16)x1;
        }
        const int kcol = kb * 32 + half * 16;  // B K-mapping
#pragma unroll
        for (int sec = 0; sec < 3; ++sec) {
          const int jr = sec * DIM + db * 16 + nn;  // output-neuron row
          v16bf bx, bh;
          const float*  xw = xwp + (size_t)jr * DIM + kcol;
          const __bf16* hw = wh_lds + jr * WP + kcol;
#pragma unroll
          for (int e = 0; e < 16; ++e) { bx[e] = (__bf16)xw[e]; bh[e] = hw[e]; }
          if (sec == 0) {
            acc_ri = __builtin_amdgcn_wmma_f32_16x16x32_bf16(false, a_x, false, bx, (short)0, acc_ri, false, false);
            acc_ri = __builtin_amdgcn_wmma_f32_16x16x32_bf16(false, a_h, false, bh, (short)0, acc_ri, false, false);
          } else if (sec == 1) {
            acc_ii = __builtin_amdgcn_wmma_f32_16x16x32_bf16(false, a_x, false, bx, (short)0, acc_ii, false, false);
            acc_ii = __builtin_amdgcn_wmma_f32_16x16x32_bf16(false, a_h, false, bh, (short)0, acc_ii, false, false);
          } else {  // newgate section kept split: resetgate scales only h_n
            acc_xn = __builtin_amdgcn_wmma_f32_16x16x32_bf16(false, a_x, false, bx, (short)0, acc_xn, false, false);
            acc_hn = __builtin_amdgcn_wmma_f32_16x16x32_bf16(false, a_h, false, bh, (short)0, acc_hn, false, false);
          }
        }
      }

      // elementwise gate update; C/D layout: M = e + 8*half, N = lane%16
#pragma unroll
      for (int e = 0; e < 8; ++e) {
        const int b = mb * 16 + e + 8 * half;
        const int d = db * 16 + nn;
        float reset = sigmoidf(acc_ri[e]);
        float inp   = sigmoidf(acc_ii[e]);
        float newg  = tanh_fast(acc_xn[e] + reset * acc_hn[e]);
        int sid2 = seq_t[b];
        float gv = g[sid2 * 3 + p];
        float conc = (gv >= EPSV) ? gv : 0.0f;
        float hold = h_lds[b * HP + d];
        float hv = hold + conc * inp * (newg - hold);
        hnew[tt][e] = hv;
        if (t == lens_lds[b] - 1)
          h_last[((size_t)p * BATCH + b) * DIM + d] = hv;
      }
    }

    __syncthreads();  // all reads of h done
#pragma unroll
    for (int tt = 0; tt < 4; ++tt) {
      const int tile = wave + tt * 32;
      const int mb = tile >> 3, db = tile & 7;
#pragma unroll
      for (int e = 0; e < 8; ++e) {
        const int b = mb * 16 + e + 8 * half;
        const int d = db * 16 + nn;
        h_lds[b * HP + d] = hnew[tt][e];
      }
    }
    __syncthreads();  // writes visible before next step's reads
  }
}

// ---------------------------------------------------------------------------
// Kernel 3: logits = sigmoid( Σ_p h_last[:,p] @ (emb*inv_norm*g[:,p])ᵀ ).
// h_last cached in LDS as bf16; each wave owns 16-item column tiles.
// ---------------------------------------------------------------------------
__global__ __launch_bounds__(256) void logits_kernel(
    const float* __restrict__ emb, const float* __restrict__ inv_norm,
    const float* __restrict__ g, const float* __restrict__ h_last,
    float* __restrict__ out) {
  __shared__ __bf16 hl[NP * BATCH * DIM];  // 192 KB
  const int tid = threadIdx.x;
  for (int i = tid; i < NP * BATCH * DIM; i += 256) hl[i] = (__bf16)h_last[i];
  __syncthreads();

  const int wave = tid >> 5, lane = tid & 31;
  const int half = lane >> 4, nn = lane & 15;
  const int n_tiles = (N_ITEMS + 15) / 16;
  const int stride = gridDim.x * 8;

  for (int it = blockIdx.x * 8 + wave; it < n_tiles; it += stride) {
    const int item = it * 16 + nn;
    const bool valid = item < N_ITEMS;
    const int isafe = valid ? item : 0;

    const float invn = inv_norm[isafe];
    const float s0 = g[isafe * 3 + 0] * invn;
    const float s1 = g[isafe * 3 + 1] * invn;
    const float s2 = g[isafe * 3 + 2] * invn;

    v16bf B0[4], B1[4], B2[4];
    const float* erow = emb + (size_t)isafe * DIM;
#pragma unroll
    for (int kb = 0; kb < 4; ++kb) {
      const int kc = kb * 32 + half * 16;
#pragma unroll
      for (int e = 0; e < 16; ++e) {
        float r = erow[kc + e];
        B0[kb][e] = (__bf16)(r * s0);
        B1[kb][e] = (__bf16)(r * s1);
        B2[kb][e] = (__bf16)(r * s2);
      }
    }

    for (int mb = 0; mb < 16; ++mb) {
      v8f acc = {};
#pragma unroll
      for (int p = 0; p < NP; ++p) {
        const __bf16* hb = hl + ((size_t)p * BATCH + mb * 16 + nn) * DIM;
#pragma unroll
        for (int kb = 0; kb < 4; ++kb) {
          v16bf a;
          const int b0 = kb * 32 + half * 16, b1 = b0 + 8;
#pragma unroll
          for (int j = 0; j < 8; ++j) { a[j] = hb[b0 + j]; a[j + 8] = hb[b1 + j]; }
          const v16bf& Bf = (p == 0) ? B0[kb] : (p == 1) ? B1[kb] : B2[kb];
          acc = __builtin_amdgcn_wmma_f32_16x16x32_bf16(false, a, false, Bf, (short)0, acc, false, false);
        }
      }
      if (valid) {
#pragma unroll
        for (int e = 0; e < 8; ++e) {
          const int b = mb * 16 + e + 8 * half;
          out[(size_t)b * N_ITEMS + item] = sigmoidf(acc[e]);
        }
      }
    }
  }
}

// ---------------------------------------------------------------------------
extern "C" void kernel_launch(void* const* d_in, const int* in_sizes, int n_in,
                              void* d_out, int out_size, void* d_ws, size_t ws_size,
                              hipStream_t stream) {
  const int*   seq     = (const int*)d_in[0];     // (100, 256)
  const int*   lens    = (const int*)d_in[1];     // (256,)
  const float* emb     = (const float*)d_in[2];   // (100001, 128)
  const float* purpose = (const float*)d_in[3];   // (3, 128)
  const float* x2h     = (const float*)d_in[4];   // (3, 384, 128)
  const float* h2h     = (const float*)d_in[5];   // (3, 384, 128)
  float* out = (float*)d_out;                     // (256, 100001)

  float* ws = (float*)d_ws;
  float* inv_norm = ws;                 // 100001 floats (padded to 100032)
  float* gbuf     = ws + 100032;        // 300003 floats (padded to 300096)
  float* h_last   = ws + 100032 + 300096;  // 3*256*128 floats

  norm_gate_kernel<<<(N_ITEMS + 7) / 8, 256, 0, stream>>>(emb, purpose, inv_norm, gbuf);
  rnn_kernel<<<NP, 1024, 0, stream>>>(seq, lens, emb, x2h, h2h, gbuf, h_last);
  // 320 blocks: one 192KB-LDS block per WGP, ~2.4 item-tiles per wave —
  // fills the machine for the dominant 19.7 GFLOP GEMM phase.
  logits_kernel<<<320, 256, 0, stream>>>(emb, inv_norm, gbuf, h_last, out);
}
